// MultiHeadAttention_25563645346642
// MI455X (gfx1250) — matrix operand order repaired and checked by reading, not compiled, not run on hardware
//
#include <hip/hip_runtime.h>
#include <hip/hip_bf16.h>

// MI455X / gfx1250, wave32. WMMA f16->f32 + TDM double-buffered K staging.
// B=8, L=1024, H=1024, NH=16, D=64.
//
// Folded algebra:  K' = keys@Wk^T + bk + posK ; V' = keys@Wv^T + bv + posV
//                  Qs = (queries@Wq^T + bq) * (1/8)      (1/sqrt(D) folded)
//                  out = softmax_causal(Qs K'^T) V'

typedef __attribute__((ext_vector_type(16))) _Float16 v16h;
typedef __attribute__((ext_vector_type(8)))  _Float16 v8h;
typedef __attribute__((ext_vector_type(4)))  _Float16 v4h;
typedef __attribute__((ext_vector_type(8)))  float    v8f;
typedef __attribute__((ext_vector_type(4)))  unsigned int u32x4;
typedef __attribute__((ext_vector_type(8)))  int      i32x8;
typedef __attribute__((ext_vector_type(4)))  int      i32x4;

#define BDIM 8
#define LDIM 1024
#define HDIM 1024
#define NHEAD 16
#define DHEAD 64

__device__ inline v16h pack16(v8h lo, v8h hi) {
  v16h r;
#pragma unroll
  for (int i = 0; i < 8; ++i) { r[i] = lo[i]; r[8 + i] = hi[i]; }
  return r;
}

// ---------------------------------------------------------------------------
// Tensor Data Mover: load one 64x64 f16 tile (rows contiguous, 128B each) from
// global into LDS, inserting 16B padding per row (pad_interval=32 DW,
// pad_amount=4 DW) -> LDS row stride 144B = conflict-friendly, 16B aligned.
// Issued by one wave; tracked with TENSORcnt.
// ---------------------------------------------------------------------------
__device__ inline void tdm_load_k_tile(const _Float16* gsrc,
                                       unsigned lds_byte_addr) {
  unsigned long long ga = (unsigned long long)(uintptr_t)gsrc;
  u32x4 g0;
  g0[0] = 1u;                                    // count=1 (valid user D#)
  g0[1] = lds_byte_addr;                         // LDS dest (bytes)
  g0[2] = (unsigned)(ga & 0xFFFFFFFFu);          // global_addr[31:0]
  g0[3] = (unsigned)((ga >> 32) & 0x01FFFFFFu)   // global_addr[56:32]
        | (2u << 30);                            // type=2 ("image")
  i32x8 g1;
  g1[0] = (int)0x07110000;   // data_size=2B | pad_enable | interval=32DW | amount=4DW
  g1[1] = (int)(64u << 16);  // tensor_dim0 = 64 elems / row
  g1[2] = (int)(64u << 16);  // tensor_dim1 = 64 rows
  g1[3] = (int)(64u << 16);  // tile_dim0   = 64
  g1[4] = 64;                // tile_dim1   = 64
  g1[5] = 64;                // tensor_dim0_stride = 64 (contiguous rows)
  g1[6] = 0; g1[7] = 0;
  i32x4 z4 = (i32x4)0;
#if __clang_major__ >= 23
  i32x8 z8 = (i32x8)0;
  __builtin_amdgcn_tensor_load_to_lds(g0, g1, z4, z4, z8, 0);
#else
  __builtin_amdgcn_tensor_load_to_lds(g0, g1, z4, z4, 0);
#endif
}

// ---------------------------------------------------------------------------
// Kernel 1: fused QKV projection (NT GEMM), f32 -> f16 head-major outputs.
// grid = (H/64, (B*L)/128, 3), block = 256 (8 waves).
// Block tile 128x64, BK=64. Wave grid 4(M) x 2(N): 8 WMMA per wave per K-iter.
// ---------------------------------------------------------------------------
__global__ __launch_bounds__(256) void proj_qkv(
    const float* __restrict__ queries, const float* __restrict__ keys,
    const float* __restrict__ posK, const float* __restrict__ posV,
    const float* __restrict__ Wq, const float* __restrict__ bq,
    const float* __restrict__ Wk, const float* __restrict__ bk,
    const float* __restrict__ Wv, const float* __restrict__ bv,
    _Float16* __restrict__ Qh, _Float16* __restrict__ Kh,
    _Float16* __restrict__ Vh) {
  const int z = blockIdx.z;
  const float* X    = (z == 0) ? queries : keys;
  const float* W    = (z == 0) ? Wq : (z == 1) ? Wk : Wv;
  const float* bias = (z == 0) ? bq : (z == 1) ? bk : bv;
  const float* pos  = (z == 0) ? nullptr : (z == 1) ? posK : posV;
  _Float16*    Out  = (z == 0) ? Qh : (z == 1) ? Kh : Vh;

  __shared__ _Float16 As[128][72];  // [m][k], row stride 144B (16B multiple)
  __shared__ _Float16 Bs[64][72];   // [n][k] (W row-major along k)

  const int tid  = threadIdx.x;
  const int lane = tid & 31;
  const int wave = tid >> 5;
  const int wm   = wave >> 1;   // 0..3
  const int wn   = wave & 1;    // 0..1
  const int lr   = lane & 15;
  const int hf   = lane >> 4;   // half-wave select
  const int m0   = blockIdx.y * 128;
  const int n0   = blockIdx.x * 64;

  v8f acc[2][2];
#pragma unroll
  for (int a = 0; a < 2; ++a)
#pragma unroll
    for (int b = 0; b < 2; ++b) acc[a][b] = (v8f)0.0f;

  for (int k0 = 0; k0 < HDIM; k0 += 64) {
    __syncthreads();
    // stage A tile: 128x64 f32 -> f16   (2048 float4 / 256 thr = 8 each)
#pragma unroll
    for (int i = 0; i < 8; ++i) {
      int linear = tid + i * 256;            // float4 units, 16 per row
      int row = linear >> 4, c4 = linear & 15;
      float4 v = *(const float4*)&X[(size_t)(m0 + row) * HDIM + k0 + c4 * 4];
      v4h h; h[0] = (_Float16)v.x; h[1] = (_Float16)v.y;
             h[2] = (_Float16)v.z; h[3] = (_Float16)v.w;
      *(v4h*)&As[row][c4 * 4] = h;
    }
    // stage B tile: 64x64 (W[n][k]) -> f16 (1024 float4, 4 each)
#pragma unroll
    for (int i = 0; i < 4; ++i) {
      int linear = tid + i * 256;
      int row = linear >> 4, c4 = linear & 15;
      float4 v = *(const float4*)&W[(size_t)(n0 + row) * HDIM + k0 + c4 * 4];
      v4h h; h[0] = (_Float16)v.x; h[1] = (_Float16)v.y;
             h[2] = (_Float16)v.z; h[3] = (_Float16)v.w;
      *(v4h*)&Bs[row][c4 * 4] = h;
    }
    __syncthreads();

#pragma unroll
    for (int kc = 0; kc < 2; ++kc) {
      v16h af[2], bf[2];
#pragma unroll
      for (int ms = 0; ms < 2; ++ms) {  // A frag: lane = M row
        int row = wm * 32 + ms * 16 + lr;
        af[ms] = pack16(*(const v8h*)&As[row][kc * 32 + hf * 8],
                        *(const v8h*)&As[row][kc * 32 + hf * 8 + 16]);
      }
#pragma unroll
      for (int ns = 0; ns < 2; ++ns) {  // B frag: lane = N col, K = hf*8 + i (+16 for i>=8)
        int row = wn * 32 + ns * 16 + lr;
        bf[ns] = pack16(*(const v8h*)&Bs[row][kc * 32 + hf * 8],
                        *(const v8h*)&Bs[row][kc * 32 + hf * 8 + 16]);
      }
#pragma unroll
      for (int ms = 0; ms < 2; ++ms)
#pragma unroll
        for (int ns = 0; ns < 2; ++ns)
          acc[ms][ns] = __builtin_amdgcn_wmma_f32_16x16x32_f16(
              false, af[ms], false, bf[ns], (short)0, acc[ms][ns], false, false);
    }
  }

  // epilogue: bias + pos + scale, convert, scatter into head-major f16
#pragma unroll
  for (int ms = 0; ms < 2; ++ms)
#pragma unroll
    for (int ns = 0; ns < 2; ++ns) {
      int gn = n0 + wn * 32 + ns * 16 + lr;
      float bv_ = bias[gn];
      int nh = gn >> 6, d = gn & 63;
#pragma unroll
      for (int j = 0; j < 8; ++j) {
        int gm = m0 + wm * 32 + ms * 16 + j + hf * 8;  // C layout: M = j + 8*hf
        float v = acc[ms][ns][j] + bv_;
        if (pos) v += pos[(size_t)gm * HDIM + gn];
        if (z == 0) v *= 0.125f;  // 1/sqrt(64)
        int bb = gm >> 10, lrow = gm & 1023;
        Out[(((size_t)bb * NHEAD + nh) * LDIM + lrow) * DHEAD + d] = (_Float16)v;
      }
    }
}

// ---------------------------------------------------------------------------
// Kernel 2: causal flash attention. grid = (L/128, B*NH), block = 256.
// Each wave owns 16 query rows; 64-key blocks; K tiles arrive via TDM,
// double-buffered so the next tile streams in during compute.
// ---------------------------------------------------------------------------
__global__ __launch_bounds__(256) void attn_kernel(
    const _Float16* __restrict__ Qh, const _Float16* __restrict__ Kh,
    const _Float16* __restrict__ Vh, float* __restrict__ out) {
  __shared__ _Float16 Ks[2][64][72];  // TDM dest: [key][d], padded to 144B rows
  __shared__ _Float16 Vt[64][72];     // [d][key] (transposed on store)
  __shared__ _Float16 Ps[128][72];    // per-wave probability staging

  const int tid  = threadIdx.x;
  const int lane = tid & 31;
  const int wave = tid >> 5;
  const int lr   = lane & 15;
  const int hf   = lane >> 4;
  const int head = blockIdx.y;            // b*NH + nh
  const int m0   = blockIdx.x * 128;
  const size_t hoff = (size_t)head * LDIM * DHEAD;

  // Q fragments (registers for whole kernel), d chunks 0-31 / 32-63
  v16h qf[2];
  {
    size_t rbase = hoff + (size_t)(m0 + wave * 16 + lr) * DHEAD;
#pragma unroll
    for (int c = 0; c < 2; ++c)
      qf[c] = pack16(*(const v8h*)&Qh[rbase + c * 32 + hf * 8],
                     *(const v8h*)&Qh[rbase + c * 32 + hf * 8 + 16]);
  }

  v8f oacc[4];
#pragma unroll
  for (int t = 0; t < 4; ++t) oacc[t] = (v8f)0.0f;
  float mrow[8], lrow[8];
#pragma unroll
  for (int j = 0; j < 8; ++j) { mrow[j] = -3.0e38f; lrow[j] = 0.0f; }

  const int nblocks = (m0 + 128) / 64;

  // prime the TDM pipeline: block 0 -> buffer 0
  if (wave == 0) tdm_load_k_tile(Kh + hoff, (unsigned)(uintptr_t)&Ks[0][0][0]);

  for (int jb = 0; jb < nblocks; ++jb) {
    const int kb  = jb * 64;
    const int cur = jb & 1;
    __syncthreads();  // compute jb-1 done: safe to overwrite Ks[cur^1], Vt

    // async: stream next K tile into the other buffer while we compute
    if (wave == 0 && jb + 1 < nblocks)
      tdm_load_k_tile(Kh + hoff + (size_t)(kb + 64) * DHEAD,
                      (unsigned)(uintptr_t)&Ks[cur ^ 1][0][0]);

    // stage V tile with transpose (512 v8h loads / 256 thr = 2 each)
#pragma unroll
    for (int i = 0; i < 2; ++i) {
      int linear = tid + i * 256;          // v8h units
      int row = linear >> 3, g = linear & 7;
      size_t src = hoff + (size_t)(kb + row) * DHEAD + g * 8;
      v8h vv = *(const v8h*)&Vh[src];
#pragma unroll
      for (int q = 0; q < 8; ++q) Vt[g * 8 + q][row] = vv[q];
      if (jb + 1 < nblocks)
        __builtin_prefetch(&Vh[src + 64 * DHEAD], 0, 1);  // global_prefetch
    }

    if (wave == 0) {
      if (jb + 1 < nblocks) __builtin_amdgcn_s_wait_tensorcnt(1);
      else                  __builtin_amdgcn_s_wait_tensorcnt(0);
    }
    __syncthreads();  // publish Ks[cur] + Vt

    // S = Q K'^T : 4 key sub-tiles x 2 d-chunks
    v8f sacc[4];
#pragma unroll
    for (int nt = 0; nt < 4; ++nt) sacc[nt] = (v8f)0.0f;
#pragma unroll
    for (int nt = 0; nt < 4; ++nt) {
      int krow = nt * 16 + lr;             // B frag: lane = key col
#pragma unroll
      for (int c = 0; c < 2; ++c) {
        v16h bf = pack16(*(const v8h*)&Ks[cur][krow][c * 32 + hf * 8],
                         *(const v8h*)&Ks[cur][krow][c * 32 + hf * 8 + 16]);
        sacc[nt] = __builtin_amdgcn_wmma_f32_16x16x32_f16(
            false, qf[c], false, bf, (short)0, sacc[nt], false, false);
      }
    }

    // causal mask (analytic; wave-uniform guard)
    if (kb + 63 > m0 + wave * 16) {
#pragma unroll
      for (int nt = 0; nt < 4; ++nt) {
        int gk = kb + nt * 16 + lr;
#pragma unroll
        for (int j = 0; j < 8; ++j) {
          int gr = m0 + wave * 16 + j + hf * 8;
          if (gk > gr) sacc[nt][j] = -1.0e30f;
        }
      }
    }

    // online softmax: rows live across the 16 lanes of each half-wave
    float rmax[8], rsum[8], alpha[8];
#pragma unroll
    for (int j = 0; j < 8; ++j) {
      float m = sacc[0][j];
#pragma unroll
      for (int nt = 1; nt < 4; ++nt) m = fmaxf(m, sacc[nt][j]);
      rmax[j] = m;
    }
#pragma unroll
    for (int j = 0; j < 8; ++j) {
#pragma unroll
      for (int d = 1; d < 16; d <<= 1)
        rmax[j] = fmaxf(rmax[j], __shfl_xor(rmax[j], d, 32));
      float mn = fmaxf(mrow[j], rmax[j]);
      alpha[j] = __expf(mrow[j] - mn);
      mrow[j] = mn;
      rsum[j] = 0.0f;
    }
#pragma unroll
    for (int nt = 0; nt < 4; ++nt)
#pragma unroll
      for (int j = 0; j < 8; ++j) {
        float p = __expf(sacc[nt][j] - mrow[j]);
        sacc[nt][j] = p;
        rsum[j] += p;
      }
#pragma unroll
    for (int j = 0; j < 8; ++j) {
#pragma unroll
      for (int d = 1; d < 16; d <<= 1) rsum[j] += __shfl_xor(rsum[j], d, 32);
      lrow[j] = lrow[j] * alpha[j] + rsum[j];
    }
#pragma unroll
    for (int t = 0; t < 4; ++t)
#pragma unroll
      for (int j = 0; j < 8; ++j) oacc[t][j] *= alpha[j];

    // C-layout -> A-layout relayout of P through this wave's private Ps rows
#pragma unroll
    for (int nt = 0; nt < 4; ++nt)
#pragma unroll
      for (int j = 0; j < 8; ++j)
        Ps[wave * 16 + j + hf * 8][nt * 16 + lr] = (_Float16)sacc[nt][j];

    // O += P V' : 2 key chunks x 4 d sub-tiles (DS in-order within a wave)
#pragma unroll
    for (int kc = 0; kc < 2; ++kc) {
      int prow = wave * 16 + lr;           // A frag: lane = query row
      v16h pf = pack16(*(const v8h*)&Ps[prow][kc * 32 + hf * 8],
                       *(const v8h*)&Ps[prow][kc * 32 + hf * 8 + 16]);
#pragma unroll
      for (int dt = 0; dt < 4; ++dt) {
        int vrow = dt * 16 + lr;           // B frag: lane = d col
        v16h vf = pack16(*(const v8h*)&Vt[vrow][kc * 32 + hf * 8],
                         *(const v8h*)&Vt[vrow][kc * 32 + hf * 8 + 16]);
        oacc[dt] = __builtin_amdgcn_wmma_f32_16x16x32_f16(
            false, pf, false, vf, (short)0, oacc[dt], false, false);
      }
    }
  }

  // normalize and write [B, L, H] f32
  const int bb = head >> 4, nh = head & 15;
  float inv[8];
#pragma unroll
  for (int j = 0; j < 8; ++j) inv[j] = 1.0f / lrow[j];
#pragma unroll
  for (int dt = 0; dt < 4; ++dt)
#pragma unroll
    for (int j = 0; j < 8; ++j) {
      int gr = m0 + wave * 16 + j + hf * 8;
      int gd = dt * 16 + lr;
      out[((size_t)bb * LDIM + gr) * HDIM + nh * DHEAD + gd] =
          oacc[dt][j] * inv[j];
    }
}

// ---------------------------------------------------------------------------
extern "C" void kernel_launch(void* const* d_in, const int* in_sizes, int n_in,
                              void* d_out, int out_size, void* d_ws,
                              size_t ws_size, hipStream_t stream) {
  (void)in_sizes; (void)n_in; (void)out_size; (void)ws_size;
  const float* queries = (const float*)d_in[0];
  const float* keys    = (const float*)d_in[1];
  // d_in[2] = attn_mask (bool) — causal mask applied analytically, unused.
  const float* posK = (const float*)d_in[3];
  const float* posV = (const float*)d_in[4];
  const float* Wq = (const float*)d_in[5];
  const float* bq = (const float*)d_in[6];
  const float* Wk = (const float*)d_in[7];
  const float* bk = (const float*)d_in[8];
  const float* Wv = (const float*)d_in[9];
  const float* bv = (const float*)d_in[10];
  float* out = (float*)d_out;

  const size_t headElems = (size_t)BDIM * NHEAD * LDIM * DHEAD;  // 8.39M
  _Float16* Qh = (_Float16*)d_ws;
  _Float16* Kh = Qh + headElems;
  _Float16* Vh = Kh + headElems;

  proj_qkv<<<dim3(HDIM / 64, (BDIM * LDIM) / 128, 3), 256, 0, stream>>>(
      queries, keys, posK, posV, Wq, bq, Wk, bk, Wv, bv, Qh, Kh, Vh);
  attn_kernel<<<dim3(LDIM / 128, BDIM * NHEAD), 256, 0, stream>>>(
      Qh, Kh, Vh, out);
}
